// LSTMDiffusion_16063177687184
// MI455X (gfx1250) — compile-verified, matrix operand order checked by reading, not run on hardware
//
#include <hip/hip_runtime.h>
#include <hip/hip_bf16.h>

// ---------------------------------------------------------------------------
// LSTM diffusion model for MI455X (gfx1250).
//  Phase A (parallel, WMMA bf16):   emb -> te MLP -> xp = x@proj^T + b + emb2
//  Phase B (parallel, WMMA bf16):   gx[l] = xp @ Wih[l]^T + bih[l] + bhh[l]
//  Phase C (sequential, f32 VALU):  persistent kernel, 4096 steps x 4 layers,
//                                   Whh@h matvec + cell update + lin output,
//                                   software grid barrier between layers.
// ---------------------------------------------------------------------------

typedef __attribute__((ext_vector_type(16))) __bf16 v16bf;
typedef __attribute__((ext_vector_type(8)))  __bf16 v8bf;
typedef __attribute__((ext_vector_type(8)))  float  v8f;

#define NSAMP 4096
#define DIM   1024
#define G4    4096   // 4 * HID

// ---------------- f32 -> bf16 conversion ----------------
__global__ void f2bf_kernel(const float* __restrict__ in, __bf16* __restrict__ out, long n) {
  long i = (long)blockIdx.x * blockDim.x + threadIdx.x;
  if (i < n) out[i] = (__bf16)in[i];
}

// ---------------- sinusoidal position embedding ----------------
__global__ void embed_kernel(const int* __restrict__ tsteps, __bf16* __restrict__ out) {
  int idx = blockIdx.x * blockDim.x + threadIdx.x;   // over NSAMP*DIM
  int i = idx >> 10;
  int j = idx & 1023;
  float tv = (float)tsteps[i];
  int jh = j & 511;
  float freq = __expf((float)jh * (-9.210340371976184f / 511.0f));
  float ang = tv * freq;
  float v = (j < 512) ? sinf(ang) : cosf(ang);
  out[idx] = (__bf16)v;
}

// ---------------- state / barrier init ----------------
__global__ void init_state_kernel(float* __restrict__ h, float* __restrict__ c,
                                  unsigned* __restrict__ bar) {
  int i = blockIdx.x * blockDim.x + threadIdx.x;
  if (i < 2048) h[i] = 0.f;
  if (i < 1024) c[i] = 0.f;
  if (i < 2)    bar[i] = 0u;
}

// ---------------- WMMA fragment load (16-bit A/B layout, ISA 7.12.2) -------
// lane m = lane&15 selects the row of the 16-wide dim; lane>>4 selects which
// 8-element K half this lane carries (K..K+7 then K+16..K+23 / +8 shifted).
__device__ __forceinline__ v16bf load_frag_bf16(const __bf16* __restrict__ tile,
                                                int ld, int k0, int lane) {
  const int m  = lane & 15;
  const int kh = (lane >> 4) & 1;
  const __bf16* p = tile + (size_t)m * ld + k0 + (kh << 3);
  v8bf lo = *(const v8bf*)(p);
  v8bf hi = *(const v8bf*)(p + 16);
  v16bf r;
#pragma unroll
  for (int i = 0; i < 8; ++i) { r[i] = lo[i]; r[i + 8] = hi[i]; }
  return r;
}

__device__ __forceinline__ v8f wmma_bf16(v16bf a, v16bf b, v8f c) {
  // (neg_a, A, neg_b, B, c_mod, C, reuse_a, reuse_b)
  return __builtin_amdgcn_wmma_f32_16x16x32_bf16(false, a, false, b, (short)0, c,
                                                 false, false);
}

// ---------------- generic bf16 GEMM:  C = act(A @ W^T + b1 [+ b2] [+ R]) ----
// A: (M,K) bf16 row-major.  W: (Ncols,K) bf16 row-major.  blockIdx.z = layer.
// Block tile 64x128 = 8 waves of 32x32.  K stepped by 32 (one WMMA per tile).
__global__ __launch_bounds__(256)
void gemm_bf16_kernel(const __bf16* __restrict__ A, const __bf16* __restrict__ W,
                      const float* __restrict__ bias1, const float* __restrict__ bias2,
                      const float* __restrict__ resid,
                      float* __restrict__ Cf, __bf16* __restrict__ Cb,
                      int M, int Ncols, int K, int act,
                      unsigned long long wStride, unsigned long long cStride,
                      unsigned long long bStride) {
  const int l = blockIdx.z;
  W += (size_t)l * wStride;
  if (bias1) bias1 += (size_t)l * bStride;
  if (bias2) bias2 += (size_t)l * bStride;
  if (Cf)    Cf    += (size_t)l * cStride;
  if (Cb)    Cb    += (size_t)l * cStride;

  const int lane  = threadIdx.x & 31;
  const int wid   = threadIdx.x >> 5;
  const int waveM = wid >> 2;          // 0..1
  const int waveN = wid & 3;           // 0..3
  const int rowBase = blockIdx.y * 64  + waveM * 32;
  const int colBase = blockIdx.x * 128 + waveN * 32;

  v8f acc[2][2];
#pragma unroll
  for (int i = 0; i < 2; ++i)
#pragma unroll
    for (int j = 0; j < 2; ++j)
#pragma unroll
      for (int v = 0; v < 8; ++v) acc[i][j][v] = 0.f;

  const __bf16* a0p = A + (size_t)(rowBase +  0) * K;
  const __bf16* a1p = A + (size_t)(rowBase + 16) * K;
  const __bf16* b0p = W + (size_t)(colBase +  0) * K;
  const __bf16* b1p = W + (size_t)(colBase + 16) * K;

  for (int k0 = 0; k0 < K; k0 += 32) {
    v16bf a0 = load_frag_bf16(a0p, K, k0, lane);
    v16bf a1 = load_frag_bf16(a1p, K, k0, lane);
    v16bf b0 = load_frag_bf16(b0p, K, k0, lane);
    v16bf b1 = load_frag_bf16(b1p, K, k0, lane);
    acc[0][0] = wmma_bf16(a0, b0, acc[0][0]);
    acc[0][1] = wmma_bf16(a0, b1, acc[0][1]);
    acc[1][0] = wmma_bf16(a1, b0, acc[1][0]);
    acc[1][1] = wmma_bf16(a1, b1, acc[1][1]);
  }

  // C/D layout: VGPR v -> row v (lanes 0-15) / row v+8 (lanes 16-31), col = lane&15
  const int lh = (lane >> 4) & 1;
  const int ln = lane & 15;
#pragma unroll
  for (int ti = 0; ti < 2; ++ti) {
#pragma unroll
    for (int tj = 0; tj < 2; ++tj) {
      const int r0 = rowBase + ti * 16 + lh * 8;
      const int c0 = colBase + tj * 16 + ln;
      float b = 0.f;
      if (bias1) b += bias1[c0];
      if (bias2) b += bias2[c0];
#pragma unroll
      for (int v = 0; v < 8; ++v) {
        const int r = r0 + v;
        float val = acc[ti][tj][v] + b;
        if (resid) val += resid[(size_t)r * Ncols + c0];
        if (act == 1) val = val / (1.f + __expf(-val));   // SiLU
        if (Cf) Cf[(size_t)r * Ncols + c0] = val;
        if (Cb) Cb[(size_t)r * Ncols + c0] = (__bf16)val;
      }
    }
  }
}

// ---------------- software grid barrier (64 resident blocks) ----------------
__device__ __forceinline__ void grid_sync(unsigned* cnt, unsigned* gen, unsigned nblk) {
  __syncthreads();
  if (threadIdx.x == 0) {
    __threadfence();
    unsigned g = *(volatile unsigned*)gen;
    if (atomicAdd(cnt, 1u) == nblk - 1u) {
      *cnt = 0u;
      __threadfence();
      atomicAdd(gen, 1u);
    } else {
      while (*(volatile unsigned*)gen == g) { __builtin_amdgcn_s_sleep(2); }
    }
    __threadfence();
  }
  __syncthreads();
}

// ---------------- sequential LSTM recurrence (persistent, 64 x 256) --------
// thread (block b, tid): hidden j = b*16 + tid/16, K-slice kp = tid&15 (64 el).
// Per layer: 4 gate partial dots vs LDS copy of h, LDS reduce, cell update,
// grid barrier.  h double-buffered (parity returns to 0 after 4 layers).
__global__ __launch_bounds__(256)
void lstm_seq_kernel(const float* __restrict__ gx,    // [4][NSAMP][4096]
                     const float* __restrict__ Whh,   // [4][4096][1024]
                     const float* __restrict__ linW,  // [1024][1024]
                     const float* __restrict__ linB,  // [1024]
                     float* __restrict__ h,           // [2][1024]
                     float* __restrict__ c,           // [1024]
                     float* __restrict__ out,         // [NSAMP][1024]
                     unsigned* __restrict__ bar) {
  __shared__ float hs[1024];
  __shared__ float red[1024];
  const int tid = threadIdx.x;
  const int j   = blockIdx.x * 16 + (tid >> 4);
  const int kp  = tid & 15;
  const int k0  = kp << 6;             // 64-element K slice
  unsigned* cnt = bar;
  unsigned* gen = bar + 1;

  for (int t = 0; t < NSAMP; ++t) {
    int cur = 0;
    for (int l = 0; l < 4; ++l) {
      for (int i = tid; i < 1024; i += 256) hs[i] = h[cur * 1024 + i];
      __syncthreads();
      const float* wb = Whh + ((size_t)l << 22);     // l*4096*1024
      const float* w0 = wb + (size_t)(j       ) * 1024 + k0;
      const float* w1 = wb + (size_t)(1024 + j) * 1024 + k0;
      const float* w2 = wb + (size_t)(2048 + j) * 1024 + k0;
      const float* w3 = wb + (size_t)(3072 + j) * 1024 + k0;
      float p0 = 0.f, p1 = 0.f, p2 = 0.f, p3 = 0.f;
#pragma unroll 8
      for (int k = 0; k < 64; ++k) {
        float hv = hs[k0 + k];
        p0 = fmaf(w0[k], hv, p0);
        p1 = fmaf(w1[k], hv, p1);
        p2 = fmaf(w2[k], hv, p2);
        p3 = fmaf(w3[k], hv, p3);
      }
      red[tid * 4 + 0] = p0; red[tid * 4 + 1] = p1;
      red[tid * 4 + 2] = p2; red[tid * 4 + 3] = p3;
      __syncthreads();
      if (kp == 0) {
        float gi = 0.f, gf = 0.f, gg = 0.f, go = 0.f;
        for (int q = 0; q < 16; ++q) {
          gi += red[(tid + q) * 4 + 0];
          gf += red[(tid + q) * 4 + 1];
          gg += red[(tid + q) * 4 + 2];
          go += red[(tid + q) * 4 + 3];
        }
        const float* gxp = gx + (((size_t)l * NSAMP + t) << 12);
        gi += gxp[j]; gf += gxp[1024 + j]; gg += gxp[2048 + j]; go += gxp[3072 + j];
        float si = 1.f / (1.f + __expf(-gi));
        float sf = 1.f / (1.f + __expf(-gf));
        float so = 1.f / (1.f + __expf(-go));
        float cn = sf * c[j] + si * tanhf(gg);
        c[j] = cn;
        h[(cur ^ 1) * 1024 + j] = so * tanhf(cn);
      }
      grid_sync(cnt, gen, gridDim.x);
      cur ^= 1;
    }
    // cur is back to 0: output row  out[t] = linW @ h + linB
    for (int i = tid; i < 1024; i += 256) hs[i] = h[i];
    __syncthreads();
    {
      const float* wr = linW + (size_t)j * 1024 + k0;
      float p = 0.f;
#pragma unroll 8
      for (int k = 0; k < 64; ++k) p = fmaf(wr[k], hs[k0 + k], p);
      red[tid] = p;
      __syncthreads();
      if (kp == 0) {
        float s = 0.f;
        for (int q = 0; q < 16; ++q) s += red[tid + q];
        out[(size_t)t * 1024 + j] = s + linB[j];
      }
    }
    __syncthreads();   // red reads done before next iteration reuses LDS
  }
}

// ---------------------------------------------------------------------------
extern "C" void kernel_launch(void* const* d_in, const int* in_sizes, int n_in,
                              void* d_out, int out_size, void* d_ws, size_t ws_size,
                              hipStream_t stream) {
  (void)in_sizes; (void)n_in; (void)out_size; (void)ws_size;
  const float* x      = (const float*)d_in[0];
  const int*   tsteps = (const int*)  d_in[1];
  const float* proj_w = (const float*)d_in[2];
  const float* proj_b = (const float*)d_in[3];
  const float* te_w1  = (const float*)d_in[4];
  const float* te_b1  = (const float*)d_in[5];
  const float* te_w2  = (const float*)d_in[6];
  const float* te_b2  = (const float*)d_in[7];
  const float* Wih    = (const float*)d_in[8];
  const float* Whh    = (const float*)d_in[9];
  const float* bih    = (const float*)d_in[10];
  const float* bhh    = (const float*)d_in[11];
  const float* lin_w  = (const float*)d_in[12];
  const float* lin_b  = (const float*)d_in[13];
  float* out = (float*)d_out;

  char* ws = (char*)d_ws;
  size_t off = 0;
  auto take = [&](size_t bytes) -> char* {
    char* p = ws + off;
    off += (bytes + 255) & ~(size_t)255;
    return p;
  };
  __bf16* x_bf    = (__bf16*)take((size_t)NSAMP * DIM * 2);
  __bf16* projw_b = (__bf16*)take((size_t)DIM * DIM * 2);
  __bf16* tew1_b  = (__bf16*)take((size_t)DIM * DIM * 2);
  __bf16* tew2_b  = (__bf16*)take((size_t)DIM * DIM * 2);
  __bf16* wih_b   = (__bf16*)take((size_t)4 * G4 * DIM * 2);
  __bf16* emb0_b  = (__bf16*)take((size_t)NSAMP * DIM * 2);
  __bf16* e1_b    = (__bf16*)take((size_t)NSAMP * DIM * 2);
  __bf16* xp_b    = (__bf16*)take((size_t)NSAMP * DIM * 2);
  float*  e2_f    = (float*) take((size_t)NSAMP * DIM * 4);
  float*  gx      = (float*) take((size_t)4 * NSAMP * G4 * 4);
  float*  hbuf    = (float*) take((size_t)2 * 1024 * 4);
  float*  cbuf    = (float*) take((size_t)1024 * 4);
  unsigned* bar   = (unsigned*)take(2 * sizeof(unsigned));

  const dim3 blk(256);
  auto blocks = [](long n) { return dim3((unsigned)((n + 255) / 256)); };

  // --- conversions / embedding / state init ---
  f2bf_kernel<<<blocks((long)NSAMP * DIM), blk, 0, stream>>>(x, x_bf, (long)NSAMP * DIM);
  f2bf_kernel<<<blocks((long)DIM * DIM),   blk, 0, stream>>>(proj_w, projw_b, (long)DIM * DIM);
  f2bf_kernel<<<blocks((long)DIM * DIM),   blk, 0, stream>>>(te_w1,  tew1_b,  (long)DIM * DIM);
  f2bf_kernel<<<blocks((long)DIM * DIM),   blk, 0, stream>>>(te_w2,  tew2_b,  (long)DIM * DIM);
  f2bf_kernel<<<blocks((long)4 * G4 * DIM), blk, 0, stream>>>(Wih, wih_b, (long)4 * G4 * DIM);
  embed_kernel<<<blocks((long)NSAMP * DIM), blk, 0, stream>>>(tsteps, emb0_b);
  init_state_kernel<<<dim3(8), blk, 0, stream>>>(hbuf, cbuf, bar);

  // --- WMMA GEMM chain ---
  // e1 = silu(emb0 @ te_w1^T + te_b1)           -> bf16
  gemm_bf16_kernel<<<dim3(DIM / 128, NSAMP / 64, 1), blk, 0, stream>>>(
      emb0_b, tew1_b, te_b1, nullptr, nullptr, nullptr, e1_b,
      NSAMP, DIM, DIM, 1, 0ull, 0ull, 0ull);
  // e2 = e1 @ te_w2^T + te_b2                   -> f32
  gemm_bf16_kernel<<<dim3(DIM / 128, NSAMP / 64, 1), blk, 0, stream>>>(
      e1_b, tew2_b, te_b2, nullptr, nullptr, e2_f, nullptr,
      NSAMP, DIM, DIM, 0, 0ull, 0ull, 0ull);
  // xp = x @ proj_w^T + proj_b + e2             -> bf16
  gemm_bf16_kernel<<<dim3(DIM / 128, NSAMP / 64, 1), blk, 0, stream>>>(
      x_bf, projw_b, proj_b, nullptr, e2_f, nullptr, xp_b,
      NSAMP, DIM, DIM, 0, 0ull, 0ull, 0ull);
  // gx[l] = xp @ Wih[l]^T + bih[l] + bhh[l]     -> f32, 4 layers via blockIdx.z
  gemm_bf16_kernel<<<dim3(G4 / 128, NSAMP / 64, 4), blk, 0, stream>>>(
      xp_b, wih_b, bih, bhh, nullptr, gx, nullptr,
      NSAMP, G4, DIM, 0,
      (unsigned long long)G4 * DIM,
      (unsigned long long)NSAMP * G4,
      (unsigned long long)G4);

  // --- sequential recurrence (persistent) ---
  lstm_seq_kernel<<<dim3(64), blk, 0, stream>>>(gx, Whh, lin_w, lin_b,
                                                hbuf, cbuf, out, bar);
}
